// InrEncoderLayer_74947179316072
// MI455X (gfx1250) — matrix operand order repaired and checked by reading, not compiled, not run on hardware
//
#include <hip/hip_runtime.h>

// ---------------------------------------------------------------------------
// Problem: B=4, N=8192, D=128, H=512
// out = x1 + FFN(LN(x1)),  x1 = x + softmax(adj) @ LN(x)
// Attention collapses to one GEMM: [8192 x 8192] x [8192 x 512] (bf16 WMMA)
// ---------------------------------------------------------------------------

typedef __attribute__((ext_vector_type(16))) __bf16 v16bf;
typedef __attribute__((ext_vector_type(8)))  float  v8f;

__device__ __forceinline__ unsigned short f2bf(float f) {
    unsigned int u = __float_as_uint(f);
    u += 0x7FFFu + ((u >> 16) & 1u);   // round-to-nearest-even
    return (unsigned short)(u >> 16);
}

// LDS-base-relative byte address of a __shared__ object (for async-to-LDS ops)
__device__ __forceinline__ unsigned ldsAddrOf(const void* p) {
    return (unsigned)(unsigned long long)
        (const __attribute__((address_space(3))) void*)p;
}

// per-lane async copy: 16 bytes global -> LDS, tracked by ASYNCcnt
__device__ __forceinline__ void asyncCopy16(unsigned lds, const void* g) {
    unsigned long long ga = (unsigned long long)g;
    asm volatile("global_load_async_to_lds_b128 %0, %1, off"
                 :: "v"(lds), "v"(ga) : "memory");
}

__device__ __forceinline__ void waitAsync0() {
    asm volatile("s_wait_asynccnt 0x0" ::: "memory");
}

// ---------------------------------------------------------------------------
// Kernel 1: row-wise softmax of adj [N][N] fp32 -> prob bf16 (row-major = A)
// ---------------------------------------------------------------------------
__global__ __launch_bounds__(256) void softmax_rows_kernel(
    const float* __restrict__ adj, unsigned short* __restrict__ prob, int N)
{
    const int row = blockIdx.x;
    const int tid = threadIdx.x;
    const float* a = adj + (size_t)row * N;

    float v[32];
    float mx = -3.4e38f;
#pragma unroll
    for (int i = 0; i < 32; ++i) {
        v[i] = a[tid + i * 256];
        mx = fmaxf(mx, v[i]);
    }

    __shared__ float red[256];
    red[tid] = mx; __syncthreads();
    for (int s = 128; s > 0; s >>= 1) {
        if (tid < s) red[tid] = fmaxf(red[tid], red[tid + s]);
        __syncthreads();
    }
    mx = red[0]; __syncthreads();

    float sum = 0.f;
#pragma unroll
    for (int i = 0; i < 32; ++i) { v[i] = __expf(v[i] - mx); sum += v[i]; }
    red[tid] = sum; __syncthreads();
    for (int s = 128; s > 0; s >>= 1) {
        if (tid < s) red[tid] += red[tid + s];
        __syncthreads();
    }
    const float inv = 1.0f / red[0];

    unsigned short* p = prob + (size_t)row * N;
#pragma unroll
    for (int i = 0; i < 32; ++i) p[tid + i * 256] = f2bf(v[i] * inv);
}

// ---------------------------------------------------------------------------
// Kernel 2: LN(x) over D, stored N-major bf16: ht[c][m], c=b*128+d, m=n
// ---------------------------------------------------------------------------
__global__ __launch_bounds__(256) void ln_to_bt_kernel(
    const float* __restrict__ x, unsigned short* __restrict__ ht)
{
    const int gw   = (blockIdx.x * blockDim.x + threadIdx.x) >> 5; // (b,n) row
    const int lane = threadIdx.x & 31;

    const float4 v = ((const float4*)(x + (size_t)gw * 128))[lane];
    float s = v.x + v.y + v.z + v.w;
    for (int o = 16; o > 0; o >>= 1) s += __shfl_xor(s, o, 32);
    const float mean = s * (1.0f / 128.0f);
    const float dx = v.x - mean, dy = v.y - mean, dz = v.z - mean, dw = v.w - mean;
    float q = dx * dx + dy * dy + dz * dz + dw * dw;
    for (int o = 16; o > 0; o >>= 1) q += __shfl_xor(q, o, 32);
    const float rstd = rsqrtf(q * (1.0f / 128.0f) + 1e-5f);

    const int b = gw >> 13;          // N = 8192
    const int n = gw & 8191;
    size_t base = (size_t)(b * 128 + lane * 4) * 8192 + n;
    ht[base]            = f2bf(dx * rstd);
    ht[base + 8192]     = f2bf(dy * rstd);
    ht[base + 2 * 8192] = f2bf(dz * rstd);
    ht[base + 3 * 8192] = f2bf(dw * rstd);
}

// ---------------------------------------------------------------------------
// Kernel 4: LN(x1) -> h1 bf16 row-major [B*N][128] (A for FFN)
// ---------------------------------------------------------------------------
__global__ __launch_bounds__(256) void ln_rows_kernel(
    const float* __restrict__ x1, unsigned short* __restrict__ h1)
{
    const int gw   = (blockIdx.x * blockDim.x + threadIdx.x) >> 5;
    const int lane = threadIdx.x & 31;

    const float4 v = ((const float4*)(x1 + (size_t)gw * 128))[lane];
    float s = v.x + v.y + v.z + v.w;
    for (int o = 16; o > 0; o >>= 1) s += __shfl_xor(s, o, 32);
    const float mean = s * (1.0f / 128.0f);
    const float dx = v.x - mean, dy = v.y - mean, dz = v.z - mean, dw = v.w - mean;
    float q = dx * dx + dy * dy + dz * dz + dw * dw;
    for (int o = 16; o > 0; o >>= 1) q += __shfl_xor(q, o, 32);
    const float rstd = rsqrtf(q * (1.0f / 128.0f) + 1e-5f);

    unsigned short* p = h1 + (size_t)gw * 128 + lane * 4;
    p[0] = f2bf(dx * rstd);
    p[1] = f2bf(dy * rstd);
    p[2] = f2bf(dz * rstd);
    p[3] = f2bf(dw * rstd);
}

// ---------------------------------------------------------------------------
// Small transpose: src fp32 [R][C] -> dst bf16 [C][R]
// ---------------------------------------------------------------------------
__global__ void transpose_to_bf16_kernel(
    const float* __restrict__ src, unsigned short* __restrict__ dst, int R, int C)
{
    int idx = blockIdx.x * 256 + threadIdx.x;
    if (idx >= R * C) return;
    int r = idx / C, c = idx % C;
    dst[(size_t)c * R + r] = f2bf(src[idx]);
}

// ---------------------------------------------------------------------------
// WMMA GEMM: C[M][Nc] = A[M][K](bf16 row-major) * Bt[Nc][K](bf16 N-major)
// BM=128, BN=64, BK=64; 8 waves, each a 32x32 C sub-tile (2x2 WMMA frags).
// Staging: double-buffered global_load_async_to_lds_b128 (ASYNCcnt).
// Epilogue modes:
//   1: outF[map(m,c)] = aux[map(m,c)] + acc      (x1 = x + attn; [B,N,D] map)
//   2: outBf[m*Nc+c]  = bf16(relu(acc + aux[c])) (hidden = relu(h1@w1+b1))
//   3: outF[m*Nc+c]  += acc + aux[c]             (out += T@w2 + b2)
// ---------------------------------------------------------------------------
#define BM 128
#define BN 64
#define BK 64
#define LDA 72
#define LDB 72

__global__ __launch_bounds__(256) void gemm_bf16_wmma_kernel(
    const unsigned short* __restrict__ A, const unsigned short* __restrict__ Bt,
    int M, int Nc, int K, int mode,
    const float* __restrict__ aux, float* __restrict__ outF,
    unsigned short* __restrict__ outBf, int batchStride)
{
    __shared__ unsigned short As[2][BM * LDA];
    __shared__ unsigned short Bs[2][BN * LDB];

    const int tid   = threadIdx.x;
    const int tileM = blockIdx.x * BM;
    const int tileN = blockIdx.y * BN;
    const int wave  = tid >> 5;
    const int lane  = tid & 31;
    const int wm    = wave & 3;   // 4 waves along M (32 rows each)
    const int wn    = wave >> 2;  // 2 waves along N (32 cols each)
    const int laneHalf = lane >> 4;
    const int laneLow  = lane & 15;

    // async-stage one BKxBM / BKxBN tile pair into LDS buffer `buf`
    auto stage = [&](int buf, int k0) {
#pragma unroll
        for (int i = 0; i < 4; ++i) {          // A: 128x64 bf16 = 1024 x b128
            int idx = tid + i * 256;
            int r = idx >> 3, c8 = idx & 7;
            asyncCopy16(ldsAddrOf(&As[buf][r * LDA + c8 * 8]),
                        A + (size_t)(tileM + r) * K + k0 + c8 * 8);
        }
#pragma unroll
        for (int i = 0; i < 2; ++i) {          // B: 64x64 bf16 = 512 x b128
            int idx = tid + i * 256;
            int r = idx >> 3, c8 = idx & 7;
            asyncCopy16(ldsAddrOf(&Bs[buf][r * LDB + c8 * 8]),
                        Bt + (size_t)(tileN + r) * K + k0 + c8 * 8);
        }
    };

    const v8f vzero = {0.f, 0.f, 0.f, 0.f, 0.f, 0.f, 0.f, 0.f};
    v8f acc[2][2];
    acc[0][0] = vzero; acc[0][1] = vzero; acc[1][0] = vzero; acc[1][1] = vzero;

    const int nkt = K / BK;
    stage(0, 0);

    for (int kt = 0; kt < nkt; ++kt) {
        const int cur = kt & 1;
        waitAsync0();          // my async copies into buffer `cur` landed
        __syncthreads();       // everyone's copies landed; prior reads retired
        if (kt + 1 < nkt) stage(cur ^ 1, (kt + 1) * BK);

        const unsigned short* as = As[cur];
        const unsigned short* bs = Bs[cur];
#pragma unroll
        for (int ks = 0; ks < 2; ++ks) {
            const int kb = ks * 32;
            v16bf afr[2], bfr[2];
            // A frag 16x32: lanes 0-15 -> K 0..7 & 16..23; lanes 16-31 -> +8
#pragma unroll
            for (int mb = 0; mb < 2; ++mb) {
                int r = wm * 32 + mb * 16 + laneLow;
                int kbase = kb + (laneHalf << 3);
                uint4* p = (uint4*)&afr[mb];
                p[0] = *(const uint4*)(&as[r * LDA + kbase]);
                p[1] = *(const uint4*)(&as[r * LDA + kbase + 16]);
            }
            // B frag 32x16: lanes 0-15 -> K 0..7 & 16..23; lanes 16-31 -> +8
#pragma unroll
            for (int nb = 0; nb < 2; ++nb) {
                int r = wn * 32 + nb * 16 + laneLow;
                int kbase = kb + (laneHalf << 3);
                uint4* p = (uint4*)&bfr[nb];
                p[0] = *(const uint4*)(&bs[r * LDB + kbase]);
                p[1] = *(const uint4*)(&bs[r * LDB + kbase + 16]);
            }
#pragma unroll
            for (int mb = 0; mb < 2; ++mb)
#pragma unroll
                for (int nb = 0; nb < 2; ++nb)
                    acc[mb][nb] = __builtin_amdgcn_wmma_f32_16x16x32_bf16(
                        false, afr[mb], false, bfr[nb],
                        (short)0, acc[mb][nb], false, false);
        }
    }

    // ---- epilogue: C frag: VGPR e -> M = e + 8*laneHalf, col = laneLow
#pragma unroll
    for (int mb = 0; mb < 2; ++mb) {
#pragma unroll
        for (int nb = 0; nb < 2; ++nb) {
            const int gmBase = tileM + wm * 32 + mb * 16 + 8 * laneHalf;
            const int gn     = tileN + wn * 32 + nb * 16 + laneLow;
#pragma unroll
            for (int e = 0; e < 8; ++e) {
                const int gm = gmBase + e;
                const float val = acc[mb][nb][e];
                if (mode == 1) {
                    size_t oi = (size_t)(gn >> 7) * batchStride
                              + (size_t)gm * 128 + (gn & 127);
                    outF[oi] = aux[oi] + val;
                } else if (mode == 2) {
                    float t = val + aux[gn];
                    t = t > 0.f ? t : 0.f;
                    outBf[(size_t)gm * Nc + gn] = f2bf(t);
                } else if (mode == 3) {
                    size_t oi = (size_t)gm * Nc + gn;
                    outF[oi] += val + aux[gn];
                } else {
                    outF[(size_t)gm * Nc + gn] = val;
                }
            }
        }
    }
}

// ---------------------------------------------------------------------------
// Launch sequence
// ---------------------------------------------------------------------------
extern "C" void kernel_launch(void* const* d_in, const int* in_sizes, int n_in,
                              void* d_out, int out_size, void* d_ws, size_t ws_size,
                              hipStream_t stream) {
    (void)in_sizes; (void)n_in; (void)out_size; (void)ws_size;

    const int Bb = 4, Nn = 8192, Dd = 128, Hh = 512;
    const float* x   = (const float*)d_in[0];
    const float* adj = (const float*)d_in[1];
    const float* w1  = (const float*)d_in[4];
    const float* b1  = (const float*)d_in[5];
    const float* w2  = (const float*)d_in[6];
    const float* b2  = (const float*)d_in[7];
    float* out = (float*)d_out;

    // workspace carve-up (bf16 buffers)
    unsigned char* ws = (unsigned char*)d_ws;
    unsigned short* prob = (unsigned short*)ws;                       // 134217728 B
    unsigned short* ht   = (unsigned short*)(ws + 134217728ull);      //   8388608 B
    unsigned short* h1   = (unsigned short*)(ws + 142606336ull);      //   8388608 B
    unsigned short* w1t  = (unsigned short*)(ws + 150994944ull);      //    131072 B
    unsigned short* w2t  = (unsigned short*)(ws + 151126016ull);      //    131072 B
    unsigned short* Tm   = (unsigned short*)(ws + 151257088ull);      //  33554432 B

    // 1) prob = softmax(adj) as bf16 (A for GEMM1)
    softmax_rows_kernel<<<Nn, 256, 0, stream>>>(adj, prob, Nn);

    // 2) ht[c][m] = LN(x)  (Bt for GEMM1)
    ln_to_bt_kernel<<<(Bb * Nn) / 8, 256, 0, stream>>>(x, ht);

    // 3) GEMM1: x1 = x + prob @ LN(x)   (M=8192, Nc=512, K=8192) -> d_out fp32
    gemm_bf16_wmma_kernel<<<dim3(Nn / BM, (Bb * Dd) / BN), 256, 0, stream>>>(
        prob, ht, Nn, Bb * Dd, Nn, /*mode=*/1, x, out, nullptr, Nn * Dd);

    // 4) h1 = LN(x1) row-major bf16 (A for GEMM2)
    ln_rows_kernel<<<(Bb * Nn) / 8, 256, 0, stream>>>(out, h1);

    // 5) w1t[h][k] = w1[k][h]  (Bt for GEMM2)
    transpose_to_bf16_kernel<<<(Dd * Hh + 255) / 256, 256, 0, stream>>>(w1, w1t, Dd, Hh);

    // 6) GEMM2: T = relu(h1 @ w1 + b1)  (M=32768, Nc=512, K=128) -> bf16
    gemm_bf16_wmma_kernel<<<dim3((Bb * Nn) / BM, Hh / BN), 256, 0, stream>>>(
        h1, w1t, Bb * Nn, Hh, Dd, /*mode=*/2, b1, nullptr, Tm, 0);

    // 7) w2t[d][h] = w2[h][d]  (Bt for GEMM3)
    transpose_to_bf16_kernel<<<(Hh * Dd + 255) / 256, 256, 0, stream>>>(w2, w2t, Hh, Dd);

    // 8) GEMM3: out += T @ w2 + b2  (M=32768, Nc=128, K=512)
    gemm_bf16_wmma_kernel<<<dim3((Bb * Nn) / BM, Dd / BN), 256, 0, stream>>>(
        Tm, w2t, Bb * Nn, Dd, Hh, /*mode=*/3, b2, out, nullptr, 0);
}
